// lstm_2405181686413
// MI455X (gfx1250) — matrix operand order repaired and checked by reading, not compiled, not run on hardware
//
#include <hip/hip_runtime.h>

// LayerNorm-LSTM fused kernel for MI455X (gfx1250, wave32, WMMA bf16).
// B=16384, T=16, I=256, H=128, G=4H=512.
//
// One block = 16 batch rows, persistent over all T=16 timesteps.
// Both GEMMs run on v_wmma_f32_16x16x32_bf16; LN stats are computed from the
// WMMA f32 accumulators via cross-lane reductions (no raw pre-LN buffer).
// Weight B-fragment loads are kept inside rolled loops (#pragma unroll 1) so
// they are NOT hoisted into registers (round-1 disasm showed scratch spills).

#define BATCH 16384
#define TT    16
#define II    256
#define HH    128
#define GG    512
#define EPSF  1e-5f
#define MTILE 16
#define NWAVES 8
#define NTPW  4   // 16-col N-tiles per wave (8 waves * 4 * 16 = 512 cols)

typedef __attribute__((ext_vector_type(16))) __bf16 bf16x16;
typedef __attribute__((ext_vector_type(8)))  float  v8f;

union FragU {
  uint4   q[2];
  bf16x16 v;
};

__device__ __forceinline__ unsigned short f32_to_bf16(float f) {
  unsigned int u = __float_as_uint(f);
  u += 0x7FFFu + ((u >> 16) & 1u);     // round-to-nearest-even
  return (unsigned short)(u >> 16);
}

__device__ __forceinline__ float sigmoidf_(float v) {
  return 1.0f / (1.0f + __expf(-v));
}

// ---------------------------------------------------------------------------
// Prep kernel: pack W_ih [512,256] and W_hh [512,128] (f32, row-major, g x k)
// into bf16 WMMA B-fragments. B-fragment K order used here (contiguous; A loads match):
//   lanes 0-15  : N = lane,    K = 0..15  (element j -> K=j)
//   lanes 16-31 : N = lane-16, K = 16..31 (element j -> K=16+j)
// Fragment (nt, ks) stored contiguously: 32 lanes x 16 bf16.
// ---------------------------------------------------------------------------
__global__ void pack_weights(const float* __restrict__ W_ih,
                             const float* __restrict__ W_hh,
                             unsigned short* __restrict__ wbuf) {
  int tid  = blockIdx.x * blockDim.x + threadIdx.x;
  int lane = tid & 31;
  int frag = tid >> 5;
  int n    = lane & 15;
  int half = lane >> 4;
  if (frag < 256) {                       // W_ih: 32 n-tiles * 8 k-steps
    int nt = frag >> 3, ks = frag & 7;
    unsigned short* dst = wbuf + ((size_t)frag * 32 + lane) * 16;
    for (int j = 0; j < 16; ++j) {
      int k = ks * 32 + half * 16 + j;
      dst[j] = f32_to_bf16(W_ih[(size_t)(nt * 16 + n) * II + k]);
    }
  } else if (frag < 384) {                // W_hh: 32 n-tiles * 4 k-steps
    int f2 = frag - 256;
    int nt = f2 >> 2, ks = f2 & 3;
    unsigned short* dst = wbuf + (size_t)256 * 512 + ((size_t)f2 * 32 + lane) * 16;
    for (int j = 0; j < 16; ++j) {
      int k = ks * 32 + half * 16 + j;
      dst[j] = f32_to_bf16(W_hh[(size_t)(nt * 16 + n) * HH + k]);
    }
  }
}

// ---------------------------------------------------------------------------
// Per-gate LayerNorm from WMMA accumulators.
// Each wave owns 4 consecutive N-tiles = 64 gate cols; a 128-col gate spans
// waves (2g, 2g+1). Per-row sums/sumsq reduced over 16 lanes (cols) per half,
// exchanged via tiny LDS arrays, then normalized values written to zg.
// Accumulator layout (16x16 f32 C/D): lanes 0-15 hold rows 0..7 (elem j = row j),
// lanes 16-31 hold rows 8..15, column = lane & 15.
// ---------------------------------------------------------------------------
__device__ __forceinline__ void ln_reduce_write(
    v8f acc[NTPW], float (*zg)[GG],
    float (*swS)[MTILE], float (*swQ)[MTILE],
    const float* __restrict__ gamma, const float* __restrict__ beta,
    const float* __restrict__ bias, bool accumulate, int wave, int lane) {
  const int half = lane >> 4;
  const int l16  = lane & 15;
  float ps[8], pq[8];
  for (int j = 0; j < 8; ++j) { ps[j] = 0.f; pq[j] = 0.f; }
  for (int nt = 0; nt < NTPW; ++nt)
    for (int j = 0; j < 8; ++j) {
      float v = acc[nt][j];
      ps[j] += v; pq[j] += v * v;
    }
  for (int m = 1; m < 16; m <<= 1)
    for (int j = 0; j < 8; ++j) {
      ps[j] += __shfl_xor(ps[j], m, 32);
      pq[j] += __shfl_xor(pq[j], m, 32);
    }
  if (l16 == 0)
    for (int j = 0; j < 8; ++j) {
      swS[wave][half * 8 + j] = ps[j];
      swQ[wave][half * 8 + j] = pq[j];
    }
  __syncthreads();
  const int p0 = wave & ~1;               // pair of waves covering this gate
  float mu[8], ri[8];
  for (int j = 0; j < 8; ++j) {
    int r   = half * 8 + j;
    float s = swS[p0][r] + swS[p0 + 1][r];
    float q = swQ[p0][r] + swQ[p0 + 1][r];
    float m_ = s * (1.f / HH);
    float v_ = q * (1.f / HH) - m_ * m_;
    mu[j] = m_;
    ri[j] = rsqrtf(v_ + EPSF);
  }
  for (int nt = 0; nt < NTPW; ++nt) {
    int col = (wave * NTPW + nt) * 16 + l16;
    float gc = gamma[col], bc = beta[col];
    float ba = bias ? bias[col] : 0.f;
    for (int j = 0; j < 8; ++j) {
      int r = half * 8 + j;
      float v = (acc[nt][j] - mu[j]) * ri[j] * gc + bc + ba;
      if (accumulate) zg[r][col] += v;
      else            zg[r][col]  = v;
    }
  }
  __syncthreads();                        // protect swS/swQ reuse
}

// ---------------------------------------------------------------------------
__global__ __launch_bounds__(256, 1)
void lstm_main(const float* __restrict__ x,   const float* __restrict__ h0,
               const float* __restrict__ c0,  const float* __restrict__ bvec,
               const float* __restrict__ g_x, const float* __restrict__ b_x,
               const float* __restrict__ g_h, const float* __restrict__ b_h,
               const float* __restrict__ g_c, const float* __restrict__ b_c,
               const unsigned short* __restrict__ wbuf,
               float* __restrict__ out) {
  __shared__ unsigned short xs[2][MTILE][II]; // 16 KB double-buffered x tile (bf16)
  __shared__ unsigned short hsh[MTILE][HH];   // 4 KB  h state (bf16)
  __shared__ float          cs[MTILE][HH];    // 8 KB  c state (f32)
  __shared__ float          zg[MTILE][GG];    // 32 KB combined gates
  __shared__ float          swS[NWAVES][MTILE];
  __shared__ float          swQ[NWAVES][MTILE];

  const int tid  = threadIdx.x;
  const int lane = tid & 31;
  const int wave = tid >> 5;
  const int half = lane >> 4;
  const int l16  = lane & 15;
  const int rowBase = blockIdx.x * MTILE;

  const unsigned short* wih = wbuf;                      // 256 frags * 512
  const unsigned short* whh = wbuf + (size_t)256 * 512;  // 128 frags * 512

  // init h, c; stage x tile for t=0
  for (int i = tid; i < MTILE * HH; i += 256) {
    int r = i >> 7, c = i & 127;
    hsh[r][c] = f32_to_bf16(h0[(size_t)(rowBase + r) * HH + c]);
    cs[r][c]  = c0[(size_t)(rowBase + r) * HH + c];
  }
  for (int k = 0; k < 4; ++k) {
    int i  = tid + k * 256;
    int r  = i >> 6;
    int c4 = (i & 63) << 2;
    const float4 v = *(const float4*)(x + ((size_t)(rowBase + r) * TT + 0) * II + c4);
    xs[0][r][c4 + 0] = f32_to_bf16(v.x);
    xs[0][r][c4 + 1] = f32_to_bf16(v.y);
    xs[0][r][c4 + 2] = f32_to_bf16(v.z);
    xs[0][r][c4 + 3] = f32_to_bf16(v.w);
  }
  __syncthreads();

  float* outH = out + (size_t)BATCH * TT * HH;

  for (int t = 0; t < TT; ++t) {
    const int cur = t & 1, nxt = cur ^ 1;

    // ---- issue global loads for tile t+1 (consumed after zh GEMM) ----
    float4 stg[4];
    if (t + 1 < TT) {
      for (int k = 0; k < 4; ++k) {
        int i  = tid + k * 256;
        int r  = i >> 6;
        int c4 = (i & 63) << 2;
        stg[k] = *(const float4*)(x + ((size_t)(rowBase + r) * TT + (t + 1)) * II + c4);
        if (t + 2 < TT)   // warm far caches for t+2 (WGP-scope prefetch)
          __builtin_prefetch(x + ((size_t)(rowBase + r) * TT + (t + 2)) * II + c4, 0, 3);
      }
    }

    // ---- recurrent GEMM: zh = h @ W_hh^T  (K=128 -> 4 wmma k-steps) ----
    {
      v8f acc[NTPW];
      for (int nt = 0; nt < NTPW; ++nt)
        for (int j = 0; j < 8; ++j) acc[nt][j] = 0.f;
#pragma unroll 1   // keep B-fragment loads inside the loop (no LICM -> no spills)
      for (int ks = 0; ks < HH / 32; ++ks) {
        const int k0 = ks * 32 + half * 16;  // A-frag: contiguous K = 16*half + j (matches B)
        FragU a;
        a.q[0] = *(const uint4*)&hsh[l16][k0];
        a.q[1] = *(const uint4*)&hsh[l16][k0 + 8];
#pragma unroll
        for (int nt = 0; nt < NTPW; ++nt) {
          const int ntg = wave * NTPW + nt;
          const bf16x16 bm =
              *(const bf16x16*)(whh + ((size_t)(ntg * (HH / 32) + ks) * 32 + lane) * 16);
          acc[nt] = __builtin_amdgcn_wmma_f32_16x16x32_bf16(
              false, a.v, false, bm, (short)0, acc[nt], false, false);
        }
      }
      // zg = LN(zh)*g_h + b_h + b
      ln_reduce_write(acc, zg, swS, swQ, g_h, b_h, bvec, false, wave, lane);
    }

    // ---- store staged tile t+1 (bf16) into the alternate buffer ----
    if (t + 1 < TT) {
      for (int k = 0; k < 4; ++k) {
        int i  = tid + k * 256;
        int r  = i >> 6;
        int c4 = (i & 63) << 2;
        xs[nxt][r][c4 + 0] = f32_to_bf16(stg[k].x);
        xs[nxt][r][c4 + 1] = f32_to_bf16(stg[k].y);
        xs[nxt][r][c4 + 2] = f32_to_bf16(stg[k].z);
        xs[nxt][r][c4 + 3] = f32_to_bf16(stg[k].w);
      }
    }

    // ---- input GEMM: zx = x_t @ W_ih^T  (K=256 -> 8 wmma k-steps) ----
    {
      v8f acc[NTPW];
      for (int nt = 0; nt < NTPW; ++nt)
        for (int j = 0; j < 8; ++j) acc[nt][j] = 0.f;
#pragma unroll 1   // keep B-fragment loads inside the loop
      for (int ks = 0; ks < II / 32; ++ks) {
        const int k0 = ks * 32 + half * 16;
        FragU a;
        a.q[0] = *(const uint4*)&xs[cur][l16][k0];
        a.q[1] = *(const uint4*)&xs[cur][l16][k0 + 8];
#pragma unroll
        for (int nt = 0; nt < NTPW; ++nt) {
          const int ntg = wave * NTPW + nt;
          const bf16x16 bm =
              *(const bf16x16*)(wih + ((size_t)(ntg * (II / 32) + ks) * 32 + lane) * 16);
          acc[nt] = __builtin_amdgcn_wmma_f32_16x16x32_bf16(
              false, a.v, false, bm, (short)0, acc[nt], false, false);
        }
      }
      // zg += LN(zx)*g_x + b_x
      ln_reduce_write(acc, zg, swS, swQ, g_x, b_x, nullptr, true, wave, lane);
    }

    // ---- cell update: c = sig(f)*c + sig(i)*tanh(g) ----
    for (int i2 = tid; i2 < MTILE * HH; i2 += 256) {
      int r = i2 >> 7, hc = i2 & 127;
      float ig = zg[r][hc];
      float fg = zg[r][HH + hc];
      float gg = zg[r][2 * HH + hc];
      cs[r][hc] = sigmoidf_(fg) * cs[r][hc] + sigmoidf_(ig) * tanhf(gg);
    }
    __syncthreads();

    // ---- cell LN + output gate: h = sig(o)*tanh(LN(c)) ; write ys (+hT) ----
    for (int rr = 0; rr < 2; ++rr) {
      int r = wave * 2 + rr;
      float cv[4];
      float s = 0.f, q = 0.f;
      for (int j = 0; j < 4; ++j) {
        cv[j] = cs[r][lane + 32 * j];
        s += cv[j];
        q += cv[j] * cv[j];
      }
      for (int m2 = 1; m2 < 32; m2 <<= 1) {
        s += __shfl_xor(s, m2, 32);
        q += __shfl_xor(q, m2, 32);
      }
      float m_ = s * (1.f / HH);
      float ri = rsqrtf(q * (1.f / HH) - m_ * m_ + EPSF);
      size_t rowG = (size_t)rowBase + r;
      for (int j = 0; j < 4; ++j) {
        int col  = lane + 32 * j;
        float cl = (cv[j] - m_) * ri * g_c[col] + b_c[col];
        float hv = sigmoidf_(zg[r][3 * HH + col]) * tanhf(cl);
        // ys / hT are streaming write-once outputs: bypass near caches
        __builtin_nontemporal_store(hv, &out[(rowG * TT + t) * HH + col]);
        hsh[r][col] = f32_to_bf16(hv);
        if (t == TT - 1)
          __builtin_nontemporal_store(hv, &outH[rowG * HH + col]);
      }
    }
    __syncthreads();
  }
}

extern "C" void kernel_launch(void* const* d_in, const int* in_sizes, int n_in,
                              void* d_out, int out_size, void* d_ws, size_t ws_size,
                              hipStream_t stream) {
  const float* x    = (const float*)d_in[0];
  const float* h0   = (const float*)d_in[1];
  const float* c0   = (const float*)d_in[2];
  const float* W_ih = (const float*)d_in[3];
  const float* W_hh = (const float*)d_in[4];
  const float* b    = (const float*)d_in[5];
  const float* g_x  = (const float*)d_in[6];
  const float* b_x  = (const float*)d_in[7];
  const float* g_h  = (const float*)d_in[8];
  const float* b_h  = (const float*)d_in[9];
  const float* g_c  = (const float*)d_in[10];
  const float* b_c  = (const float*)d_in[11];
  float* out = (float*)d_out;
  unsigned short* wbuf = (unsigned short*)d_ws;   // 384 KB packed bf16 weights

  pack_weights<<<48, 256, 0, stream>>>(W_ih, W_hh, wbuf);
  lstm_main<<<BATCH / MTILE, 256, 0, stream>>>(
      x, h0, c0, b, g_x, b_x, g_h, b_h, g_c, b_c, wbuf, out);
}
